// ParallelTransformerBlock_68006512165477
// MI455X (gfx1250) — compile-verified, matrix operand order checked by reading, not run on hardware
//
#include <hip/hip_runtime.h>

// ---------------------------------------------------------------------------
// CDNA5 (gfx1250) parallel transformer block, bf16 WMMA pipeline with
// async global->LDS staging (ASYNCcnt) and double-buffered GEMM tiles.
// ---------------------------------------------------------------------------

typedef __attribute__((ext_vector_type(16))) __bf16 v16bf;
typedef __attribute__((ext_vector_type(8)))  float  v8f;

#define WMMA_BF16(a, b, c) \
  __builtin_amdgcn_wmma_f32_16x16x32_bf16(false, (a), false, (b), (short)0, (c), false, false)

static constexpr int kDim       = 2048;
static constexpr int kSeq       = 2048;
static constexpr int kBatch     = 2;
static constexpr int kHeads     = 16;
static constexpr int kDh        = 64;
static constexpr int kAttnInner = 1024;   // 16*64
static constexpr int kFF        = 8192;   // DIM*4
static constexpr int kFusedN    = 17536;  // 1024 + 64 + 64 + 16384
static constexpr int kRows      = kBatch * kSeq; // 4096
static constexpr int kQEnd      = 1024;
static constexpr int kKEnd      = 1088;
static constexpr int kVEnd      = 1152;

// ---------------------- async global->LDS (ASYNCcnt) -----------------------
// GVS mode: mem = SGPR64 + VGPR32 + ioffset; VDST VGPR holds the LDS byte
// address. Tracked by ASYNCcnt; completion enforced with s_wait_asynccnt.
__device__ __forceinline__ void async_ld_b128(void* lds_dst, const void* gbase,
                                              unsigned goff_bytes) {
  unsigned lds_off = (unsigned)(uintptr_t)lds_dst;  // addr[31:0] == LDS offset
  asm volatile("global_load_async_to_lds_b128 %0, %1, %2"
               :
               : "v"(lds_off), "v"(goff_bytes), "s"(gbase)
               : "memory");
}
__device__ __forceinline__ void wait_async0() {
  asm volatile("s_wait_asynccnt 0x0" ::: "memory");
}

// -------------------------- fragment loader --------------------------------
// 16-bit A/B operand layout (ISA 7.12.2): lane&15 selects the row (A: M, B: N),
// upper half-wave covers K offsets {8..15, 24..31}. Two contiguous 16B loads.
__device__ __forceinline__ v16bf frag_ld(const __bf16* base, int stride, int kbase) {
  const int lane = threadIdx.x & 31;
  const __bf16* p = base + (size_t)(lane & 15) * stride + kbase + ((lane >> 4) << 3);
  union { uint4 u[2]; v16bf v; } t;
  t.u[0] = *reinterpret_cast<const uint4*>(p);
  t.u[1] = *reinterpret_cast<const uint4*>(p + 16);
  return t.v;
}

__device__ __forceinline__ float half_max(float v) {
  v = fmaxf(v, __shfl_xor(v, 1, 32));
  v = fmaxf(v, __shfl_xor(v, 2, 32));
  v = fmaxf(v, __shfl_xor(v, 4, 32));
  v = fmaxf(v, __shfl_xor(v, 8, 32));
  return v;
}
__device__ __forceinline__ float half_sum(float v) {
  v += __shfl_xor(v, 1, 32);
  v += __shfl_xor(v, 2, 32);
  v += __shfl_xor(v, 4, 32);
  v += __shfl_xor(v, 8, 32);
  return v;
}

// ------------------------ weight prep (fold LoRA) --------------------------
__global__ void k_prep_fused(const float* __restrict__ wf,
                             const float* __restrict__ qa, const float* __restrict__ qb,
                             const float* __restrict__ ka, const float* __restrict__ kb,
                             const float* __restrict__ va, const float* __restrict__ vb,
                             __bf16* __restrict__ out) {
  size_t i = (size_t)blockIdx.x * 256 + threadIdx.x;
  const size_t total = (size_t)kDim * kFusedN;
  if (i >= total) return;
  const int d = (int)(i / kFusedN);
  const int j = (int)(i % kFusedN);
  float v = wf[i];
  if (j < kQEnd) {
#pragma unroll
    for (int r = 0; r < 8; ++r) v += qa[d * 8 + r] * qb[r * kAttnInner + j];
  } else if (j < kKEnd) {
    const int jj = j - kQEnd;
#pragma unroll
    for (int r = 0; r < 8; ++r) v += ka[d * 8 + r] * kb[r * kDh + jj];
  } else if (j < kVEnd) {
    const int jj = j - kKEnd;
#pragma unroll
    for (int r = 0; r < 8; ++r) v += va[d * 8 + r] * vb[r * kDh + jj];
  }
  out[i] = (__bf16)v;
}

__global__ void k_prep_attn_out(const float* __restrict__ w,
                                const float* __restrict__ oa, const float* __restrict__ ob,
                                __bf16* __restrict__ out) {
  size_t i = (size_t)blockIdx.x * 256 + threadIdx.x;
  const size_t total = (size_t)kAttnInner * kDim;
  if (i >= total) return;
  const int r0 = (int)(i / kDim);
  const int j  = (int)(i % kDim);
  float v = w[i];
#pragma unroll
  for (int r = 0; r < 8; ++r) v += oa[r0 * 8 + r] * ob[r * kDim + j];
  out[i] = (__bf16)v;
}

__global__ void k_prep_ff_out(const float* __restrict__ w, __bf16* __restrict__ out) {
  size_t i = (size_t)blockIdx.x * 256 + threadIdx.x;
  const size_t total = (size_t)kFF * kDim;
  if (i >= total) return;
  out[i] = (__bf16)w[i];
}

// ------------------------------- layernorm ---------------------------------
__global__ __launch_bounds__(256) void k_layernorm(const float* __restrict__ x,
                                                   const float* __restrict__ gamma,
                                                   __bf16* __restrict__ xn) {
  const int row = blockIdx.x;
  const float* xr = x + (size_t)row * kDim;
  __shared__ float s1[256], s2[256];
  float a = 0.f, b = 0.f;
  for (int c = threadIdx.x; c < kDim; c += 256) {
    const float v = xr[c];
    a += v; b += v * v;
  }
  s1[threadIdx.x] = a; s2[threadIdx.x] = b;
  __syncthreads();
  for (int st = 128; st > 0; st >>= 1) {
    if (threadIdx.x < (unsigned)st) {
      s1[threadIdx.x] += s1[threadIdx.x + st];
      s2[threadIdx.x] += s2[threadIdx.x + st];
    }
    __syncthreads();
  }
  const float mu   = s1[0] * (1.f / kDim);
  const float var  = s2[0] * (1.f / kDim) - mu * mu;
  const float rstd = rsqrtf(var + 1e-5f);
  for (int c = threadIdx.x; c < kDim; c += 256)
    xn[(size_t)row * kDim + c] = (__bf16)((xr[c] - mu) * rstd * gamma[c]);
}

// ------------------------------ tiled GEMM ---------------------------------
// C[M,N] = A[M,K] @ B[K,N], bf16 in, f32 accum. 128x128x32 block tile,
// 8 waves (2x4), each wave 64x32 = 4x2 WMMA accumulators.
// Double-buffered LDS: A tiles copied with global_load_async_to_lds_b128
// (overlapped with WMMA on the other buffer); B tiles transposed via regs+ds.
// MODE: 0 -> bf16 store, 1 -> f32 store, 2 -> f32 accumulate (+=).
static constexpr int kLdsStride = 40;  // 32 + 8 pad: 80B rows, 16B aligned, conflict-free

template <int MODE>
__global__ __launch_bounds__(256) void k_gemm(const __bf16* __restrict__ A,
                                              const __bf16* __restrict__ B,
                                              void* __restrict__ Cv,
                                              int M, int N, int K) {
  __shared__ __attribute__((aligned(16))) __bf16 As[2][128 * kLdsStride];
  __shared__ __attribute__((aligned(16))) __bf16 Bs[2][128 * kLdsStride];
  const int tid  = threadIdx.x;
  const int lane = tid & 31;
  const int wave = tid >> 5;
  const int m0 = blockIdx.y * 128;
  const int n0 = blockIdx.x * 128;
  const int wm = (wave >> 2) * 64;
  const int wn = (wave & 3) * 32;

  // Per-thread staging roles (two b128 transfers each for A and for B).
  const int arow0 = tid >> 1, aseg0 = (tid & 1) << 4;          // i = tid      (row, 2 segs merged)
  // A: 512 b128 ops: i in {tid, tid+256}: row = i>>2, seg = (i&3)*8
  auto stage_a = [&](int buf, int kt) {
#pragma unroll
    for (int j = 0; j < 2; ++j) {
      const int i = tid + j * 256;
      const int row = i >> 2, seg = (i & 3) << 3;
      async_ld_b128(&As[buf][row * kLdsStride + seg], A,
                    (unsigned)((((size_t)(m0 + row) * K) + kt + seg) * 2));
    }
  };
  uint4 bv[2];
  auto load_b = [&](int kt) {
#pragma unroll
    for (int j = 0; j < 2; ++j) {
      const int i = tid + j * 256;
      const int kk = i >> 4, nc = (i & 15) << 3;
      bv[j] = *reinterpret_cast<const uint4*>(B + (size_t)(kt + kk) * N + n0 + nc);
    }
  };
  auto store_b = [&](int buf) {
#pragma unroll
    for (int j = 0; j < 2; ++j) {
      const int i = tid + j * 256;
      const int kk = i >> 4, nc = (i & 15) << 3;
      const __bf16* e = reinterpret_cast<const __bf16*>(&bv[j]);
#pragma unroll
      for (int t = 0; t < 8; ++t) Bs[buf][(nc + t) * kLdsStride + kk] = e[t];
    }
  };
  (void)arow0; (void)aseg0;

  v8f zero = {};
  v8f acc[4][2];
#pragma unroll
  for (int i = 0; i < 4; ++i)
#pragma unroll
    for (int j = 0; j < 2; ++j) acc[i][j] = zero;

  // Prologue: stage tile 0 into buffer 0.
  stage_a(0, 0);
  load_b(0);
  store_b(0);
  wait_async0();
  __syncthreads();

  int cur = 0;
  for (int kt = 0; kt < K; kt += 32) {
    const bool has_next = (kt + 32) < K;
    if (has_next) {          // overlap next tile's copy with this tile's math
      stage_a(cur ^ 1, kt + 32);
      load_b(kt + 32);
    }

    v16bf af[4], bfg[2];
#pragma unroll
    for (int mt = 0; mt < 4; ++mt)
      af[mt] = frag_ld(&As[cur][(wm + mt * 16) * kLdsStride], kLdsStride, 0);
#pragma unroll
    for (int nt = 0; nt < 2; ++nt)
      bfg[nt] = frag_ld(&Bs[cur][(wn + nt * 16) * kLdsStride], kLdsStride, 0);
#pragma unroll
    for (int mt = 0; mt < 4; ++mt)
#pragma unroll
      for (int nt = 0; nt < 2; ++nt)
        acc[mt][nt] = WMMA_BF16(af[mt], bfg[nt], acc[mt][nt]);

    if (has_next) {
      store_b(cur ^ 1);
      wait_async0();
    }
    __syncthreads();
    cur ^= 1;
  }

  // C/D layout: n = lane&15, rows r + 8*(lane>=16).
  const int nn  = lane & 15;
  const int mhi = (lane >> 4) << 3;
#pragma unroll
  for (int mt = 0; mt < 4; ++mt)
#pragma unroll
    for (int nt = 0; nt < 2; ++nt)
#pragma unroll
      for (int r = 0; r < 8; ++r) {
        const int m = m0 + wm + mt * 16 + mhi + r;
        const int n = n0 + wn + nt * 16 + nn;
        const size_t idx = (size_t)m * N + n;
        const float v = acc[mt][nt][r];
        if (MODE == 0)      ((__bf16*)Cv)[idx] = (__bf16)v;
        else if (MODE == 1) ((float*)Cv)[idx]  = v;
        else                ((float*)Cv)[idx] += v;
      }
}

// ----------------------------- RoPE + repack -------------------------------
__global__ void k_rope(const __bf16* __restrict__ fused,
                       __bf16* __restrict__ qout,   // [B,H,N,64]
                       __bf16* __restrict__ kout,   // [B,N,64]
                       __bf16* __restrict__ vout) { // [B,N,64]
  size_t i = (size_t)blockIdx.x * 256 + threadIdx.x;
  const size_t total = (size_t)kRows * kVEnd;
  if (i >= total) return;
  const int row = (int)(i / kVEnd);
  const int c   = (int)(i % kVEnd);
  const int b = row >> 11, n = row & (kSeq - 1);
  const __bf16* fr = fused + (size_t)row * kFusedN;

  if (c < kQEnd) {
    const int h = c >> 6, d = c & 63;
    const float t  = (float)fr[c] * 0.125f;  // 64^-0.5
    const int dp   = (d < 32) ? d + 32 : d - 32;
    const float tp = (float)fr[h * 64 + dp] * 0.125f;
    const float ang = (float)n * __expf(-(float)(d & 31) * (9.210340372f / 32.f));
    const float cs = __cosf(ang), sn = __sinf(ang);
    const float o = (d < 32) ? (t * cs - tp * sn) : (t * cs + tp * sn);
    qout[((size_t)(b * kHeads + h) * kSeq + n) * kDh + d] = (__bf16)o;
  } else if (c < kKEnd) {
    const int d = c - kQEnd;
    const float t  = (float)fr[kQEnd + d];
    const int dp   = (d < 32) ? d + 32 : d - 32;
    const float tp = (float)fr[kQEnd + dp];
    const float ang = (float)n * __expf(-(float)(d & 31) * (9.210340372f / 32.f));
    const float cs = __cosf(ang), sn = __sinf(ang);
    const float o = (d < 32) ? (t * cs - tp * sn) : (t * cs + tp * sn);
    kout[((size_t)b * kSeq + n) * kDh + d] = (__bf16)o;
  } else {
    const int d = c - kKEnd;
    vout[((size_t)b * kSeq + n) * kDh + d] = fr[kKEnd + d];
  }
}

// ---------------------- flash attention (causal, MQA) ----------------------
// Block: 128 threads = 4 waves; each wave owns 16 q rows; block covers 64 rows.
// K tiles copied with async global->LDS; V transposed at stage time via regs.
__global__ __launch_bounds__(128) void k_attn(const __bf16* __restrict__ Q,
                                              const __bf16* __restrict__ Kr,
                                              const __bf16* __restrict__ V,
                                              __bf16* __restrict__ O) { // [B,N,1024]
  __shared__ __attribute__((aligned(16))) __bf16 Ks[32 * 64];
  __shared__ __attribute__((aligned(16))) __bf16 Vt[64 * 32];
  __shared__ __attribute__((aligned(16))) __bf16 Ps[4][16 * 32];
  const int tid  = threadIdx.x;
  const int lane = tid & 31;
  const int w    = tid >> 5;
  const int qt = blockIdx.x, h = blockIdx.y, b = blockIdx.z;
  const int qr0 = qt * 64 + w * 16;

  const __bf16* Qg = Q + ((size_t)(b * kHeads + h) * kSeq + qr0) * kDh;
  const v16bf qf0 = frag_ld(Qg, kDh, 0);
  const v16bf qf1 = frag_ld(Qg, kDh, 32);

  v8f zero = {};
  v8f accO[4];
#pragma unroll
  for (int i = 0; i < 4; ++i) accO[i] = zero;
  float mrow[8], lrow[8];
#pragma unroll
  for (int r = 0; r < 8; ++r) { mrow[r] = -1e30f; lrow[r] = 0.f; }

  const int nn  = lane & 15;
  const int mhi = (lane >> 4) << 3;
  const int jblocks = qt * 2 + 2;  // causal: only kv blocks up to this q tile
  const __bf16* Kg = Kr + (size_t)b * kSeq * kDh;
  const __bf16* Vg = V  + (size_t)b * kSeq * kDh;

  for (int jb = 0; jb < jblocks; ++jb) {
    const int j0 = jb * 32;
    // Stage K tile 32x64 via async global->LDS (straight copy).
#pragma unroll
    for (int j = 0; j < 2; ++j) {
      const int i = tid + j * 128;
      const int r = i >> 3, cseg = (i & 7) << 3;
      async_ld_b128(&Ks[r * 64 + cseg], Kg,
                    (unsigned)((((size_t)(j0 + r)) * kDh + cseg) * 2));
    }
    // Stage V transposed: Vt[d][j] so PV's B operand is b128-loadable.
#pragma unroll
    for (int j = 0; j < 2; ++j) {
      const int i = tid + j * 128;
      const int r = i >> 3, cseg = (i & 7) << 3;
      const uint4 dv = *reinterpret_cast<const uint4*>(Vg + (size_t)(j0 + r) * kDh + cseg);
      const __bf16* e = reinterpret_cast<const __bf16*>(&dv);
#pragma unroll
      for (int t = 0; t < 8; ++t) Vt[(cseg + t) * 32 + r] = e[t];
    }
    wait_async0();
    __syncthreads();

    // S = q @ k^T for two 16-col kv tiles.
    v8f s[2];
#pragma unroll
    for (int nt = 0; nt < 2; ++nt) {
      const v16bf k0 = frag_ld(&Ks[(nt * 16) * 64], 64, 0);
      const v16bf k1 = frag_ld(&Ks[(nt * 16) * 64], 64, 32);
      v8f z = zero;
      z = WMMA_BF16(qf0, k0, z);
      z = WMMA_BF16(qf1, k1, z);
      s[nt] = z;
    }

    // Causal mask + online softmax (per-row stats via half-wave shuffles).
    float p[2][8];
#pragma unroll
    for (int nt = 0; nt < 2; ++nt)
#pragma unroll
      for (int r = 0; r < 8; ++r) {
        const int jc = j0 + nt * 16 + nn;
        const int qr = qr0 + mhi + r;
        p[nt][r] = (jc <= qr) ? s[nt][r] : -1e30f;
      }
#pragma unroll
    for (int r = 0; r < 8; ++r) {
      const float mx = half_max(fmaxf(p[0][r], p[1][r]));
      const float mn = fmaxf(mrow[r], mx);
      const float sc = __expf(mrow[r] - mn);
      const float p0 = __expf(p[0][r] - mn);
      const float p1 = __expf(p[1][r] - mn);
      lrow[r] = lrow[r] * sc + half_sum(p0 + p1);
      mrow[r] = mn;
      p[0][r] = p0; p[1][r] = p1;
#pragma unroll
      for (int n4 = 0; n4 < 4; ++n4) accO[n4][r] *= sc;
    }

    // P -> per-wave LDS (D layout -> A layout round trip), then O += P @ V.
    __bf16* Pw = Ps[w];
#pragma unroll
    for (int nt = 0; nt < 2; ++nt)
#pragma unroll
      for (int r = 0; r < 8; ++r)
        Pw[(mhi + r) * 32 + nt * 16 + nn] = (__bf16)p[nt][r];
    // Same-wave LDS ops are in-order on CDNA5; RAW through Pw is safe.
    const v16bf pf = frag_ld(Pw, 32, 0);
#pragma unroll
    for (int n4 = 0; n4 < 4; ++n4) {
      const v16bf vf = frag_ld(&Vt[(n4 * 16) * 32], 32, 0);
      accO[n4] = WMMA_BF16(pf, vf, accO[n4]);
    }
    __syncthreads();
  }

  // Normalize and store as out[b, n, h*64 + d].
#pragma unroll
  for (int n4 = 0; n4 < 4; ++n4)
#pragma unroll
    for (int r = 0; r < 8; ++r) {
      const float v = accO[n4][r] * (1.f / lrow[r]);
      const int row = qr0 + mhi + r;
      const int col = h * kDh + n4 * 16 + nn;
      O[((size_t)b * kSeq + row) * kAttnInner + col] = (__bf16)v;
    }
}

// -------------------------------- SwiGLU -----------------------------------
__global__ void k_swiglu(const __bf16* __restrict__ fused, __bf16* __restrict__ hout) {
  size_t i = (size_t)blockIdx.x * 256 + threadIdx.x;
  const size_t total = (size_t)kRows * kFF;
  if (i >= total) return;
  const int row = (int)(i / kFF);
  const int c   = (int)(i % kFF);
  const __bf16* fr = fused + (size_t)row * kFusedN + kVEnd;
  const float xf = (float)fr[c];
  const float g  = (float)fr[kFF + c];
  const float sig = 1.f / (1.f + __expf(-g));
  hout[i] = (__bf16)(g * sig * xf);
}

// ------------------------------ launcher -----------------------------------
extern "C" void kernel_launch(void* const* d_in, const int* in_sizes, int n_in,
                              void* d_out, int out_size, void* d_ws, size_t ws_size,
                              hipStream_t stream) {
  (void)in_sizes; (void)n_in; (void)out_size; (void)ws_size;
  const float* x     = (const float*)d_in[0];
  const float* gamma = (const float*)d_in[1];
  const float* wfu   = (const float*)d_in[2];
  const float* wao   = (const float*)d_in[3];
  const float* wff   = (const float*)d_in[4];
  const float* qa = (const float*)d_in[5];
  const float* qb = (const float*)d_in[6];
  const float* ka = (const float*)d_in[7];
  const float* kb = (const float*)d_in[8];
  const float* va = (const float*)d_in[9];
  const float* vb = (const float*)d_in[10];
  const float* oa = (const float*)d_in[11];
  const float* ob = (const float*)d_in[12];
  float* out = (float*)d_out;

  char* ws = (char*)d_ws;
  size_t off = 0;
  auto carve = [&](size_t bytes) -> char* {
    char* p = ws + off;
    off += (bytes + 255) & ~(size_t)255;
    return p;
  };
  __bf16* wfu_bf  = (__bf16*)carve((size_t)kDim * kFusedN * 2);
  __bf16* wao_bf  = (__bf16*)carve((size_t)kAttnInner * kDim * 2);
  __bf16* wff_bf  = (__bf16*)carve((size_t)kFF * kDim * 2);
  __bf16* xn      = (__bf16*)carve((size_t)kRows * kDim * 2);
  __bf16* fused_b = (__bf16*)carve((size_t)kRows * kFusedN * 2);
  __bf16* q_buf   = (__bf16*)carve((size_t)kBatch * kHeads * kSeq * kDh * 2);
  __bf16* k_buf   = (__bf16*)carve((size_t)kBatch * kSeq * kDh * 2);
  __bf16* v_buf   = (__bf16*)carve((size_t)kBatch * kSeq * kDh * 2);
  __bf16* attn_c  = (__bf16*)carve((size_t)kRows * kAttnInner * 2);
  __bf16* h_buf   = (__bf16*)carve((size_t)kRows * kFF * 2);

  // 1) Weight prep (fold LoRA, cast bf16).
  {
    const size_t t = (size_t)kDim * kFusedN;
    k_prep_fused<<<(unsigned)((t + 255) / 256), 256, 0, stream>>>(wfu, qa, qb, ka, kb, va, vb, wfu_bf);
  }
  {
    const size_t t = (size_t)kAttnInner * kDim;
    k_prep_attn_out<<<(unsigned)((t + 255) / 256), 256, 0, stream>>>(wao, oa, ob, wao_bf);
  }
  {
    const size_t t = (size_t)kFF * kDim;
    k_prep_ff_out<<<(unsigned)((t + 255) / 256), 256, 0, stream>>>(wff, wff_bf);
  }

  // 2) LayerNorm.
  k_layernorm<<<kRows, 256, 0, stream>>>(x, gamma, xn);

  // 3) Fused projection GEMM: [4096,2048] x [2048,17536] -> bf16.
  k_gemm<0><<<dim3(kFusedN / 128, kRows / 128), 256, 0, stream>>>(
      xn, wfu_bf, (void*)fused_b, kRows, kFusedN, kDim);

  // 4) RoPE + repack q/k/v.
  {
    const size_t t = (size_t)kRows * kVEnd;
    k_rope<<<(unsigned)((t + 255) / 256), 256, 0, stream>>>(fused_b, q_buf, k_buf, v_buf);
  }

  // 5) Flash attention.
  k_attn<<<dim3(kSeq / 64, kHeads, kBatch), 128, 0, stream>>>(q_buf, k_buf, v_buf, attn_c);

  // 6) SwiGLU.
  {
    const size_t t = (size_t)kRows * kFF;
    k_swiglu<<<(unsigned)((t + 255) / 256), 256, 0, stream>>>(fused_b, h_buf);
  }

  // 7) attn-out GEMM writes d_out; 8) ff-out GEMM accumulates into d_out.
  k_gemm<1><<<dim3(kDim / 128, kRows / 128), 256, 0, stream>>>(
      attn_c, wao_bf, (void*)out, kRows, kDim, kAttnInner);
  k_gemm<2><<<dim3(kDim / 128, kRows / 128), 256, 0, stream>>>(
      h_buf, wff_bf, (void*)out, kRows, kDim, kFF);
}
